// FeatureCubeMLP_18176301597180
// MI455X (gfx1250) — matrix operand order repaired and checked by reading, not compiled, not run on hardware
//
#include <hip/hip_runtime.h>

typedef __attribute__((ext_vector_type(16))) _Float16 v16h;
typedef __attribute__((ext_vector_type(8)))  _Float16 v8h;
typedef __attribute__((ext_vector_type(8)))  float    v8f;

constexpr int D = 128;               // padded feature / hidden dim
constexpr int WAVES = 8;             // 256 threads / wave32
constexpr int PTS_PER_WAVE = 16;     // one WMMA M-tile per wave
constexpr int PTS_PER_BLOCK = WAVES * PTS_PER_WAVE;   // 128

// ---- A fragment: 16x32 f16 tile of row-major feat[16][128] at K-offset k0 ----
// ISA 7.12.2: lanes 0-15 hold row m=lane, K = {k0..k0+7, k0+16..k0+23};
//             lanes 16-31 hold row m=lane-16, K = {k0+8..k0+15, k0+24..k0+31}.
__device__ __forceinline__ v16h load_A(const _Float16* feat, int lane, int k0) {
    int m  = lane & 15;
    int kb = (lane >> 4) << 3;            // 0 or 8
    const _Float16* p = feat + m * D + k0 + kb;
    v8h lo = *(const v8h*)(p);            // K = k0+kb .. +7
    v8h hi = *(const v8h*)(p + 16);       // K = k0+kb+16 .. +23
    v16h a;
#pragma unroll
    for (int i = 0; i < 8; ++i) { a[i] = lo[i]; a[i + 8] = hi[i]; }
    return a;
}

// ---- B fragment: 32x16 f16 tile of W^T, W stored row-major [out][128] in LDS.
// B[k][n] = W[n0+n][k0+k]; lanes 0-15: n=lane, K = {k0..k0+7, k0+16..k0+23};
// lanes 16-31: n=lane-16, K = {k0+8..k0+15, k0+24..k0+31}.
__device__ __forceinline__ v16h load_B(const _Float16* W, int lane, int n0, int k0) {
    int n  = lane & 15;
    int kb = (lane >> 4) << 3;            // 0 or 8
    const _Float16* p = W + (n0 + n) * D + k0 + kb;
    v8h lo = *(const v8h*)(p);
    v8h hi = *(const v8h*)(p + 16);
    v16h b;
#pragma unroll
    for (int i = 0; i < 8; ++i) { b[i] = lo[i]; b[i + 8] = hi[i]; }
    return b;
}

// Dense layer: feat(16x128,f16,LDS) @ Ws^T(+bias) -> overwrite feat (optionally ReLU).
template <bool RELU>
__device__ __forceinline__ void dense128(_Float16* feat, const _Float16* Ws,
                                         const float* bs, int lane) {
    v16h A[4];
#pragma unroll
    for (int k = 0; k < 4; ++k) A[k] = load_A(feat, lane, k * 32);
    const int n = lane & 15, g = lane >> 4;
#pragma unroll
    for (int t = 0; t < 8; ++t) {
        const int n0 = t * 16;
        float bv = bs[n0 + n];
        v8f acc;
#pragma unroll
        for (int r = 0; r < 8; ++r) acc[r] = bv;
#pragma unroll
        for (int k = 0; k < 4; ++k) {
            v16h B = load_B(Ws, lane, n0, k * 32);
            acc = __builtin_amdgcn_wmma_f32_16x16x32_f16(false, A[k], false, B,
                                                         (short)0, acc, false, false);
        }
        // D layout: lane n = lane&15, VGPR r -> row m = r + 8*(lane>>4)
#pragma unroll
        for (int r = 0; r < 8; ++r) {
            float v = acc[r];
            if (RELU) v = fmaxf(v, 0.f);
            feat[(r + 8 * g) * D + n0 + n] = (_Float16)v;
        }
    }
}

__global__ __launch_bounds__(256)
void fcmlp_kernel(const float* __restrict__ x,
                  const float* __restrict__ pxy, const float* __restrict__ pyz,
                  const float* __restrict__ pxz,
                  const float* __restrict__ W1, const float* __restrict__ b1,
                  const float* __restrict__ W2, const float* __restrict__ b2,
                  const float* __restrict__ W3, const float* __restrict__ b3,
                  float* __restrict__ out, int N) {
    __shared__ _Float16 W1s[128 * D];
    __shared__ _Float16 W2s[128 * D];
    __shared__ _Float16 W3s[16 * D];
    __shared__ float b1s[128], b2s[128], b3s[16];
    __shared__ _Float16 featS[WAVES][PTS_PER_WAVE * D];

    const int tid = threadIdx.x;

    // ---- stage weights (f16) + biases into LDS, pad d_in 123->128, W3 rows 4->16 ----
    for (int i = tid; i < 128 * D; i += 256) {
        int r = i >> 7, c = i & 127;
        W1s[i] = (_Float16)(c < 123 ? W1[r * 123 + c] : 0.f);
        W2s[i] = (_Float16)(W2[i]);
    }
    for (int i = tid; i < 16 * D; i += 256) {
        int r = i >> 7, c = i & 127;
        W3s[i] = (_Float16)(r < 4 ? W3[r * 128 + c] : 0.f);
    }
    if (tid < 128) { b1s[tid] = b1[tid]; b2s[tid] = b2[tid]; }
    if (tid < 16)  { b3s[tid] = (tid < 4) ? b3[tid] : 0.f; }
    __syncthreads();

    const int wave = tid >> 5;
    const int lane = tid & 31;
    _Float16* feat = featS[wave];
    const int pbase = blockIdx.x * PTS_PER_BLOCK + wave * PTS_PER_WAVE;

    // ---- feature gather: 2 lanes per point, 16 channels each ----
    {
        const int lp   = lane >> 1;
        const int half = lane & 1;
        int p = min(pbase + lp, N - 1);
        float cx = x[p * 3 + 0], cy = x[p * 3 + 1], cz = x[p * 3 + 2];
        const float* planes[3] = {pxy, pyz, pxz};
        float hc[3] = {cy * 255.f, cy * 255.f, cz * 255.f};
        float wc[3] = {cx * 255.f, cz * 255.f, cx * 255.f};
#pragma unroll
        for (int pl = 0; pl < 3; ++pl) {
            float h = fminf(fmaxf(hc[pl], 0.f), 255.f);
            float w = fminf(fmaxf(wc[pl], 0.f), 255.f);
            float h0f = floorf(h), w0f = floorf(w);
            int h0 = (int)h0f, w0 = (int)w0f;
            int h1 = min(h0 + 1, 255), w1 = min(w0 + 1, 255);
            float dh = h - h0f, dw = w - w0f;
            float w00 = (1.f - dh) * (1.f - dw);
            float w01 = (1.f - dh) * dw;
            float w10 = dh * (1.f - dw);
            float w11 = dh * dw;
            const float* r0 = planes[pl] + h0 * 256;
            const float* r1 = planes[pl] + h1 * 256;
            const int cbase = half * 16;
#pragma unroll 4
            for (int c = 0; c < 16; ++c) {
                const float* q0 = r0 + (cbase + c) * 65536;
                const float* q1 = r1 + (cbase + c) * 65536;
                float v = q0[w0] * w00 + q0[w1] * w01 + q1[w0] * w10 + q1[w1] * w11;
                feat[lp * D + pl * 32 + cbase + c] = (_Float16)v;
            }
        }
    }

    // ---- positional encoding: lanes 0-15, one point each; cols 96..122, pad 123..127 ----
    if (lane < 16) {
        int p = min(pbase + lane, N - 1);
        float co[3] = {x[p * 3 + 0], x[p * 3 + 1], x[p * 3 + 2]};
        _Float16* row = feat + lane * D + 96;
        row[0] = (_Float16)co[0]; row[1] = (_Float16)co[1]; row[2] = (_Float16)co[2];
#pragma unroll
        for (int f = 0; f < 4; ++f) {
            float fr = (float)(1 << f) * 6.28318530717958647692f;
#pragma unroll
            for (int d = 0; d < 3; ++d) {
                float s = co[d] * fr;
                row[3 + f * 3 + d]      = (_Float16)__sinf(s);
                row[3 + 12 + f * 3 + d] = (_Float16)__cosf(s);
            }
        }
#pragma unroll
        for (int c = 123; c < 128; ++c) feat[lane * D + c] = (_Float16)0.f;
    }
    __syncthreads();

    // ---- MLP via WMMA ----
    dense128<true>(feat, W1s, b1s, lane);   // h1 = relu(feat @ W1^T + b1)
    __syncthreads();
    dense128<true>(feat, W2s, b2s, lane);   // h2 = relu(h1 @ W2^T + b2)
    __syncthreads();

    // layer 3: one 16-wide output tile (cols 0..3 valid)
    {
        v16h A[4];
#pragma unroll
        for (int k = 0; k < 4; ++k) A[k] = load_A(feat, lane, k * 32);
        const int n = lane & 15, g = lane >> 4;
        float bv = b3s[n];
        v8f acc;
#pragma unroll
        for (int r = 0; r < 8; ++r) acc[r] = bv;
#pragma unroll
        for (int k = 0; k < 4; ++k) {
            v16h B = load_B(W3s, lane, 0, k * 32);
            acc = __builtin_amdgcn_wmma_f32_16x16x32_f16(false, A[k], false, B,
                                                         (short)0, acc, false, false);
        }
        if (n < 4) {
#pragma unroll
            for (int r = 0; r < 8; ++r) {
                int p = pbase + r + 8 * g;
                if (p < N) out[p * 4 + n] = acc[r];
            }
        }
    }
}

extern "C" void kernel_launch(void* const* d_in, const int* in_sizes, int n_in,
                              void* d_out, int out_size, void* d_ws, size_t ws_size,
                              hipStream_t stream) {
    const float* x   = (const float*)d_in[0];
    const float* pxy = (const float*)d_in[1];
    const float* pyz = (const float*)d_in[2];
    const float* pxz = (const float*)d_in[3];
    const float* W1  = (const float*)d_in[4];
    const float* b1  = (const float*)d_in[5];
    const float* W2  = (const float*)d_in[6];
    const float* b2  = (const float*)d_in[7];
    const float* W3  = (const float*)d_in[8];
    const float* b3  = (const float*)d_in[9];
    float* out = (float*)d_out;

    const int N = in_sizes[0] / 3;                       // 500000
    const int blocks = (N + PTS_PER_BLOCK - 1) / PTS_PER_BLOCK;
    hipLaunchKernelGGL(fcmlp_kernel, dim3(blocks), dim3(256), 0, stream,
                       x, pxy, pyz, pxz, W1, b1, W2, b2, W3, b3, out, N);
}
